// LSTMTagger_8693013807720
// MI455X (gfx1250) — matrix operand order repaired and checked by reading, not compiled, not run on hardware
//
#include <hip/hip_runtime.h>
#include <hip/hip_bf16.h>

// ---------------------------------------------------------------- constants
#define S_LEN    4096
#define LCH      16
#define CHAR_DIM 64
#define CHAR_HID 128
#define N_DIM    256
#define N_HID    512
#define N_TAG    64
#define FEAT     (N_DIM + CHAR_HID)   // 384
#define CG       (4 * CHAR_HID)       // 512 char gates
#define WG       (4 * N_HID)          // 2048 word gates
#define NBLK     16                   // persistent WGs for recurrence
#define KPB      (N_HID / NBLK)       // 32 hidden channels per WG

typedef __attribute__((ext_vector_type(16))) __bf16 bf16x16;
typedef __attribute__((ext_vector_type(8)))  __bf16 bf16x8;
typedef __attribute__((ext_vector_type(8)))  float  f32x8;
typedef __attribute__((ext_vector_type(4)))  unsigned tdm_g0_t;
typedef __attribute__((ext_vector_type(8)))  int      tdm_g1_t;
typedef __attribute__((ext_vector_type(4)))  int      tdm_g4_t;

// ---------------------------------------------------------------- helpers
__device__ __forceinline__ __bf16 f2bf(float f) {
  unsigned u = __builtin_bit_cast(unsigned, f);
  unsigned r = u + 0x7FFFu + ((u >> 16) & 1u);          // round-to-nearest-even
  unsigned short h = (unsigned short)(r >> 16);
  return __builtin_bit_cast(__bf16, h);
}
__device__ __forceinline__ float sigm(float x) { return 1.0f / (1.0f + __expf(-x)); }

// A fragment (16x32 bf16, row-major source, ld elements). Per ISA 7.12.2:
// lane l holds row m0 + (l&15); element e -> K = kb + (e/8)*16 + (l/16)*8 + (e%8).
// ld==0 replicates one row across all 16 M rows (GEMV trick).
__device__ __forceinline__ bf16x16 load_A(const __bf16* X, int ld, int m0, int kb, int lane) {
  const __bf16* row = X + (size_t)(m0 + (lane & 15)) * (size_t)ld;
  int k0 = kb + ((lane >> 4) << 3);
  bf16x8 lo = *(const bf16x8*)(row + k0);
  bf16x8 hi = *(const bf16x8*)(row + k0 + 16);
  bf16x16 a;
#pragma unroll
  for (int i = 0; i < 8; ++i) { a[i] = lo[i]; a[8 + i] = hi[i]; }
  return a;
}

// B fragment (32x16 bf16) from row-major W (G x K): B[k][n] = W[g0+n][k].
// lane l holds column n = l&15; element e -> K = kb + (e/8)*16 + (l/16)*8 + (e%8).
__device__ __forceinline__ bf16x16 load_B(const __bf16* W, int ld, int g0, int kb, int lane) {
  const __bf16* row = W + (size_t)(g0 + (lane & 15)) * (size_t)ld + kb + ((lane >> 4) << 3);
  bf16x8 lo = *(const bf16x8*)(row);
  bf16x8 hi = *(const bf16x8*)(row + 16);
  bf16x16 w;
#pragma unroll
  for (int i = 0; i < 8; ++i) { w[i] = lo[i]; w[8 + i] = hi[i]; }
  return w;
}

// ---------------------------------------------------------------- kernel 1: weight prep
__global__ void prep_kernel(const float* __restrict__ cWih, const float* __restrict__ cWhh,
                            const float* __restrict__ cbih, const float* __restrict__ cbhh,
                            const float* __restrict__ Wih,  const float* __restrict__ Whh,
                            const float* __restrict__ bih,  const float* __restrict__ bhh,
                            __bf16* cWih_bf, __bf16* cWhh_bf, float* cbias,
                            __bf16* Wih_bf, float* wbias, __bf16* WhhPack) {
  int tid = blockIdx.x * blockDim.x + threadIdx.x;
  int nt  = gridDim.x * blockDim.x;
  for (int i = tid; i < CG * CHAR_DIM; i += nt) cWih_bf[i] = f2bf(cWih[i]);
  for (int i = tid; i < CG * CHAR_HID; i += nt) cWhh_bf[i] = f2bf(cWhh[i]);
  for (int i = tid; i < CG; i += nt)            cbias[i]   = cbih[i] + cbhh[i];
  for (int i = tid; i < WG * FEAT; i += nt)     Wih_bf[i]  = f2bf(Wih[i]);
  for (int i = tid; i < WG; i += nt)            wbias[i]   = bih[i] + bhh[i];
  // Pack Whh rows so each recurrence WG owns [i|f|g|o] rows for its 32 channels.
  for (int i = tid; i < WG * N_HID; i += nt) {
    int g = i / N_HID, c = i % N_HID;
    int q = g >> 9, k = g & 511;            // gate section, hidden channel
    int b = k >> 5, r = (q << 5) + (k & 31);
    WhhPack[((size_t)(b * 128 + r)) * N_HID + c] = f2bf(Whh[i]);
  }
}

// ---------------------------------------------------------------- kernel 2: word embedding gather
__global__ void embed_kernel(const float* __restrict__ word_emb, const int* __restrict__ x,
                             __bf16* feat) {
  int tid = blockIdx.x * blockDim.x + threadIdx.x;
  int nt  = gridDim.x * blockDim.x;
  for (int i = tid; i < S_LEN * N_DIM; i += nt) {
    int s = i >> 8, d = i & (N_DIM - 1);
    feat[(size_t)s * FEAT + d] = f2bf(word_emb[(size_t)x[s] * N_DIM + d]);
  }
}

// ---------------------------------------------------------------- kernel 3: char LSTM (WMMA)
__global__ void __launch_bounds__(256)
char_lstm_kernel(const float* __restrict__ char_emb, const int* __restrict__ chars,
                 const int* __restrict__ lens,
                 const __bf16* __restrict__ cWih_bf, const __bf16* __restrict__ cWhh_bf,
                 const float* __restrict__ cbias, __bf16* feat) {
  __shared__ __bf16 sX[16 * CHAR_DIM];
  __shared__ __bf16 sH[16 * CHAR_HID];
  __shared__ float  sC[16 * CHAR_HID];
  __shared__ float  sG[16 * CG];

  int tid  = threadIdx.x;
  int lane = tid & 31, wave = tid >> 5;
  int m    = tid >> 4;          // word row 0..15
  int sub  = tid & 15;
  int w0   = blockIdx.x * 16;
  int mylen = lens[w0 + m];

  for (int i = tid; i < 16 * CHAR_HID; i += 256) { sH[i] = f2bf(0.0f); sC[i] = 0.0f; }
  __syncthreads();

  for (int t = 0; t < LCH; ++t) {
    // gather char embeddings -> LDS (bf16)
    int cid = chars[(w0 + m) * LCH + t];
#pragma unroll
    for (int j = 0; j < 4; ++j) {
      int d = sub * 4 + j;
      sX[m * CHAR_DIM + d] = f2bf(char_emb[(size_t)cid * CHAR_DIM + d]);
    }
    __syncthreads();

    // gates(16x512) = sX @ cWih^T + sH @ cWhh^T + bias  (32 N-tiles, 4 per wave)
#pragma unroll
    for (int j = 0; j < 4; ++j) {
      int g0 = (wave * 4 + j) * 16;
      float b = cbias[g0 + (lane & 15)];
      f32x8 acc;
#pragma unroll
      for (int r = 0; r < 8; ++r) acc[r] = b;
#pragma unroll
      for (int kb = 0; kb < CHAR_DIM; kb += 32) {
        bf16x16 a = load_A(sX, CHAR_DIM, 0, kb, lane);
        bf16x16 w = load_B(cWih_bf, CHAR_DIM, g0, kb, lane);
        acc = __builtin_amdgcn_wmma_f32_16x16x32_bf16(false, a, false, w, (short)0, acc, false, false);
      }
#pragma unroll
      for (int kb = 0; kb < CHAR_HID; kb += 32) {
        bf16x16 a = load_A(sH, CHAR_HID, 0, kb, lane);
        bf16x16 w = load_B(cWhh_bf, CHAR_HID, g0, kb, lane);
        acc = __builtin_amdgcn_wmma_f32_16x16x32_bf16(false, a, false, w, (short)0, acc, false, false);
      }
#pragma unroll
      for (int r = 0; r < 8; ++r)
        sG[(r + (lane >> 4) * 8) * CG + g0 + (lane & 15)] = acc[r];
    }
    __syncthreads();

    // pointwise cell with length mask (h,c frozen past sequence end)
    if (t < mylen) {
#pragma unroll
      for (int j = 0; j < 8; ++j) {
        int k = sub * 8 + j;
        float gi = sG[m * CG + k];
        float gf = sG[m * CG + CHAR_HID + k];
        float gg = sG[m * CG + 2 * CHAR_HID + k];
        float go = sG[m * CG + 3 * CHAR_HID + k];
        float c  = sigm(gf) * sC[m * CHAR_HID + k] + sigm(gi) * tanhf(gg);
        float h  = sigm(go) * tanhf(c);
        sC[m * CHAR_HID + k] = c;
        sH[m * CHAR_HID + k] = f2bf(h);
      }
    }
    __syncthreads();
  }

  // final h -> feat columns [256,384)
#pragma unroll
  for (int j = 0; j < 8; ++j) {
    int k = sub * 8 + j;
    feat[(size_t)(w0 + m) * FEAT + N_DIM + k] = sH[m * CHAR_HID + k];
  }
}

// ---------------------------------------------------------------- kernel 4: word input projection GEMM
__global__ void __launch_bounds__(256)
proj_kernel(const __bf16* __restrict__ feat, const __bf16* __restrict__ Wih_bf,
            const float* __restrict__ wbias, float* __restrict__ gih) {
  int lane = threadIdx.x & 31, wave = threadIdx.x >> 5;
  int m0 = blockIdx.x * 16;
  int g0 = blockIdx.y * 128 + wave * 16;
  float b = wbias[g0 + (lane & 15)];
  f32x8 acc;
#pragma unroll
  for (int r = 0; r < 8; ++r) acc[r] = b;
#pragma unroll
  for (int kb = 0; kb < FEAT; kb += 32) {
    bf16x16 a = load_A(feat, FEAT, m0, kb, lane);
    bf16x16 w = load_B(Wih_bf, FEAT, g0, kb, lane);
    acc = __builtin_amdgcn_wmma_f32_16x16x32_bf16(false, a, false, w, (short)0, acc, false, false);
  }
#pragma unroll
  for (int r = 0; r < 8; ++r)
    gih[(size_t)(m0 + r + (lane >> 4) * 8) * WG + g0 + (lane & 15)] = acc[r];
}

// ---------------------------------------------------------------- kernel 5: serial word recurrence
// 16 persistent WGs; WG b owns channels [32b,32b+32) and the 128 packed Whh rows.
// The 128KB Whh slice is staged into LDS ONCE via the Tensor Data Mover (TDM),
// then re-read 4096 times from LDS instead of L2. h replicated across 16 WMMA
// M-rows => GEMV via WMMA, two independent accumulator chains for ILP.
// Double-buffered h + monotonic device-scope barrier per step.
__global__ void __launch_bounds__(256)
recur_kernel(const float* __restrict__ gih, const __bf16* __restrict__ WhhPack,
             __bf16* hbuf, float* __restrict__ hs, unsigned* cnt) {
  extern __shared__ __bf16 sW[];               // 128 x 512 bf16 = 128 KB (dynamic LDS)
  __shared__ float sG[128];
  __shared__ float sC[KPB];
  int tid = threadIdx.x, lane = tid & 31, wave = tid >> 5;
  int b   = blockIdx.x;
  const __bf16* Wb = WhhPack + (size_t)b * 128 * N_HID;
  int r0   = wave * 16;                          // this wave's packed-row tile
  int rcol = r0 + (lane & 15);                   // packed row of this lane's gate
  int gorig = (rcol >> 5) * N_HID + b * KPB + (rcol & 31);  // original gate index

  // ---- TDM: DMA the 2D tile (512 cols x 128 rows, 2B elems) global -> LDS.
  // D# per ISA ch.8: group0 = {count, lds_addr, global_addr, type=2},
  // group1 = {data_size=2B, tensor_dim0=512, tensor_dim1=128, tile 512x128, stride0=512}.
  if (tid < 32) {                                // wave 0 issues the descriptor
    unsigned lds_base = (unsigned)(size_t)(void*)sW;   // low 32 bits = LDS byte offset
    unsigned long long ga = (unsigned long long)(size_t)Wb;
    tdm_g0_t g0v;
    g0v[0] = 1u;                                           // count=1 (valid descriptor)
    g0v[1] = lds_base;                                     // lds_addr
    g0v[2] = (unsigned)ga;                                 // global_addr[31:0]
    g0v[3] = (unsigned)((ga >> 32) & 0x01FFFFFFu) | 0x80000000u; // addr[56:32] | type=2
    tdm_g1_t g1v;
    g1v[0] = (int)(1u << 16);                              // data_size=1 -> 2 bytes
    g1v[1] = (int)(512u << 16);                            // tensor_dim0 = 512
    g1v[2] = (int)(128u << 16);                            // tensor_dim1 = 128
    g1v[3] = (int)(512u << 16);                            // tile_dim0 = 512
    g1v[4] = 128;                                          // tile_dim1 = 128
    g1v[5] = 512;                                          // tensor_dim0_stride = 512
    g1v[6] = 0;
    g1v[7] = 0;
    tdm_g4_t gz; gz[0] = 0; gz[1] = 0; gz[2] = 0; gz[3] = 0;
#if __clang_major__ >= 23
    tdm_g1_t gz8; for (int i = 0; i < 8; ++i) gz8[i] = 0;
    __builtin_amdgcn_tensor_load_to_lds(g0v, g1v, gz, gz, gz8, 0);
#else
    __builtin_amdgcn_tensor_load_to_lds(g0v, g1v, gz, gz, 0);
#endif
    __builtin_amdgcn_s_wait_tensorcnt((short)0);           // TENSORcnt -> 0
  }
  if (tid < KPB) sC[tid] = 0.0f;
  __syncthreads();                               // LDS weights visible to all waves

  for (int t = 0; t < S_LEN; ++t) {
    const __bf16* hb_r = hbuf + (size_t)(t & 1) * N_HID;        // h_{t}
    __bf16*       hb_w = hbuf + (size_t)((t + 1) & 1) * N_HID;  // h_{t+1}

    if (t + 1 < S_LEN)
      __builtin_prefetch(gih + (size_t)(t + 1) * WG + ((b * 256 + tid * 8) & (WG - 1)), 0, 1);

    float cinit = gih[(size_t)t * WG + gorig];   // precomputed x-projection (+bias)
    f32x8 acc0, acc1;
#pragma unroll
    for (int r = 0; r < 8; ++r) { acc0[r] = cinit; acc1[r] = 0.0f; }
#pragma unroll
    for (int kb = 0; kb < N_HID; kb += 64) {     // two independent WMMA chains
      bf16x16 a0 = load_A(hb_r, /*ld=*/0, 0, kb, lane);        // replicated-row h
      bf16x16 w0 = load_B(sW, N_HID, r0, kb, lane);            // weights from LDS
      acc0 = __builtin_amdgcn_wmma_f32_16x16x32_bf16(false, a0, false, w0, (short)0, acc0, false, false);
      bf16x16 a1 = load_A(hb_r, /*ld=*/0, 0, kb + 32, lane);
      bf16x16 w1 = load_B(sW, N_HID, r0, kb + 32, lane);
      acc1 = __builtin_amdgcn_wmma_f32_16x16x32_bf16(false, a1, false, w1, (short)0, acc1, false, false);
    }
    if (lane < 16) sG[r0 + lane] = acc0[0] + acc1[0];  // M=0 row = VGPR0, lanes 0..15
    __syncthreads();

    if (tid < KPB) {                             // cell update for our 32 channels
      float gi = sG[tid], gf = sG[32 + tid], gg = sG[64 + tid], go = sG[96 + tid];
      float c  = sigm(gf) * sC[tid] + sigm(gi) * tanhf(gg);
      float h  = sigm(go) * tanhf(c);
      sC[tid] = c;
      hs[(size_t)t * N_HID + b * KPB + tid] = h;
      hb_w[b * KPB + tid] = f2bf(h);
    }
    __syncthreads();

    if (tid == 0) {                              // device-scope split barrier
      __hip_atomic_fetch_add(cnt, 1u, __ATOMIC_RELEASE, __HIP_MEMORY_SCOPE_AGENT);
      unsigned target = (unsigned)NBLK * (unsigned)(t + 1);
      while (__hip_atomic_load(cnt, __ATOMIC_ACQUIRE, __HIP_MEMORY_SCOPE_AGENT) < target)
        __builtin_amdgcn_s_sleep(2);
    }
    __syncthreads();
  }
}

// ---------------------------------------------------------------- kernel 6: classifier head + log_softmax
__global__ void __launch_bounds__(64)
head_kernel(const float* __restrict__ hs, const float* __restrict__ W1,
            const float* __restrict__ b1, float* __restrict__ out) {
  __shared__ float red[64];
  int s = blockIdx.x, j = threadIdx.x;
  const float* h = hs + (size_t)s * N_HID;
  const float* w = W1 + (size_t)j * N_HID;
  float acc = b1[j];
#pragma unroll 8
  for (int k = 0; k < N_HID; ++k) acc = fmaf(h[k], w[k], acc);

  red[j] = acc;
  __syncthreads();
  for (int off = 32; off > 0; off >>= 1) {
    if (j < off) red[j] = fmaxf(red[j], red[j + off]);
    __syncthreads();
  }
  float mx = red[0];
  __syncthreads();
  red[j] = __expf(acc - mx);
  __syncthreads();
  for (int off = 32; off > 0; off >>= 1) {
    if (j < off) red[j] += red[j + off];
    __syncthreads();
  }
  float lse = mx + __logf(red[0]);
  out[(size_t)s * N_TAG + j] = acc - lse;
}

// ---------------------------------------------------------------- launch
extern "C" void kernel_launch(void* const* d_in, const int* in_sizes, int n_in,
                              void* d_out, int out_size, void* d_ws, size_t ws_size,
                              hipStream_t stream) {
  const float* char_emb = (const float*)d_in[0];
  const float* cWih     = (const float*)d_in[1];
  const float* cWhh     = (const float*)d_in[2];
  const float* cbih     = (const float*)d_in[3];
  const float* cbhh     = (const float*)d_in[4];
  const float* word_emb = (const float*)d_in[5];
  const float* Wih      = (const float*)d_in[6];
  const float* Whh      = (const float*)d_in[7];
  const float* bih      = (const float*)d_in[8];
  const float* bhh      = (const float*)d_in[9];
  const float* W1       = (const float*)d_in[10];
  const float* b1       = (const float*)d_in[11];
  const int*   x        = (const int*)d_in[12];
  const int*   chars    = (const int*)d_in[13];
  const int*   lens     = (const int*)d_in[14];

  // workspace layout (bump allocator, 256B aligned)
  char* p = (char*)d_ws;
  unsigned* cnt  = (unsigned*)p;            // [0,64)    barrier counter
  __bf16*   hbuf = (__bf16*)(p + 64);       // [64,2112) double-buffered h (2*512 bf16)
  size_t off = 4096;
  auto alloc = [&](size_t bytes) -> char* {
    char* q = p + off;
    off = (off + bytes + 255) & ~(size_t)255;
    return q;
  };
  __bf16* cWih_bf = (__bf16*)alloc((size_t)CG * CHAR_DIM * 2);
  __bf16* cWhh_bf = (__bf16*)alloc((size_t)CG * CHAR_HID * 2);
  float*  cbias   = (float*) alloc((size_t)CG * 4);
  __bf16* Wih_bf  = (__bf16*)alloc((size_t)WG * FEAT * 2);
  float*  wbias   = (float*) alloc((size_t)WG * 4);
  __bf16* WhhPack = (__bf16*)alloc((size_t)WG * N_HID * 2);
  __bf16* feat    = (__bf16*)alloc((size_t)S_LEN * FEAT * 2);
  float*  gih     = (float*) alloc((size_t)S_LEN * WG * 4);
  float*  hsbuf   = (float*) alloc((size_t)S_LEN * N_HID * 4);

  // allow 128KB dynamic LDS for the recurrence kernel (320KB/WGP available)
  (void)hipFuncSetAttribute((const void*)recur_kernel,
                            hipFuncAttributeMaxDynamicSharedMemorySize, 128 * 1024);

  hipMemsetAsync(d_ws, 0, 4096, stream);    // zero barrier counter + h double-buffer

  prep_kernel<<<1024, 256, 0, stream>>>(cWih, cWhh, cbih, cbhh, Wih, Whh, bih, bhh,
                                        cWih_bf, cWhh_bf, cbias, Wih_bf, wbias, WhhPack);
  embed_kernel<<<1024, 256, 0, stream>>>(word_emb, x, feat);
  char_lstm_kernel<<<S_LEN / 16, 256, 0, stream>>>(char_emb, chars, lens,
                                                   cWih_bf, cWhh_bf, cbias, feat);
  proj_kernel<<<dim3(S_LEN / 16, WG / 128), 256, 0, stream>>>(feat, Wih_bf, wbias, gih);
  recur_kernel<<<NBLK, 256, 128 * 1024, stream>>>(gih, WhhPack, hbuf, hsbuf, cnt);
  head_kernel<<<S_LEN, 64, 0, stream>>>(hsbuf, W1, b1, (float*)d_out);
}
